// LocalAttentionUnFold_10144712753205
// MI455X (gfx1250) — matrix operand order repaired and checked by reading, not compiled, not run on hardware
//
#include <hip/hip_runtime.h>
#include <hip/hip_bf16.h>

// Problem constants (B=1)
#define S_LEN 2048
#define E_DIM 1024
#define NH    16
#define HD    64
#define SPAD  (S_LEN + 64)      // 32 zero rows of padding on each side
#define QK_SCALE 0.125f         // 64^-0.5

typedef __attribute__((ext_vector_type(16))) __bf16 v16bf;
typedef __attribute__((ext_vector_type(8)))  __bf16 v8bf;
typedef __attribute__((ext_vector_type(8)))  float  v8f;

// 16-byte global -> LDS async copy (ASYNCcnt-tracked, no VGPR round trip).
__device__ __forceinline__ void async_cp16(void* lds, const void* g) {
  unsigned loff =
      (unsigned)(size_t)(__attribute__((address_space(3))) void*)lds;
  asm volatile("global_load_async_to_lds_b128 %0, %1, off"
               :: "v"(loff), "v"(g) : "memory");
}
__device__ __forceinline__ void wait_async0() {
  asm volatile("s_wait_asynccnt 0x0" ::: "memory");
}

// A-matrix (16x32 bf16) K-index for fragment element e (0..15), lane half
__device__ __forceinline__ int kaIdx(int e, int half) {
  return (e < 8) ? (8 * half + e) : (16 + 8 * half + (e - 8));
}

// Build a 16x32 A-fragment from a bf16 row: two contiguous 16B chunks.
__device__ __forceinline__ v16bf load_afrag(const __bf16* row, int half) {
  v8bf lo = *(const v8bf*)(row + 16 * half);
  v8bf hi = *(const v8bf*)(row + 16 * half + 8);
  return __builtin_shufflevector(lo, hi, 0, 1, 2, 3, 4, 5, 6, 7,
                                 8, 9, 10, 11, 12, 13, 14, 15);
}

// ---------------------------------------------------------------------------
// Prep: fp32 -> bf16 conversion (native cvt), 8 elements/thread.
// ---------------------------------------------------------------------------
__global__ __launch_bounds__(256)
void cvt_bf16_kernel(const float* __restrict__ in, __bf16* __restrict__ out,
                     int n) {
  int i = (blockIdx.x * 256 + threadIdx.x) * 8;
  if (i >= n) return;
  v8bf o;
#pragma unroll
  for (int e = 0; e < 8; ++e) o[e] = (__bf16)in[i + e];
  *(v8bf*)(out + i) = o;
}

// Zero the 32-row halo around each head's K rows and V-transposed columns.
__global__ __launch_bounds__(256)
void zero_pad_kernel(__bf16* __restrict__ kgp, __bf16* __restrict__ vtp) {
  int t = blockIdx.x * 256 + threadIdx.x;   // 65536 threads
  if (t >= NH * 64 * HD) return;
  int h = t / (64 * HD);
  int r = (t / HD) & 63;
  int d = t & 63;
  int row = (r < 32) ? r : (S_LEN + r);     // 0..31 and S+32..S+63
  kgp[h * SPAD * HD + row * HD + d] = (__bf16)0.f;
  vtp[h * HD * SPAD + d * SPAD + row] = (__bf16)0.f;
}

// ---------------------------------------------------------------------------
// Kernel 1: fused Q/K/V projection (bf16 WMMA, f32 accumulate).
// 256 threads / 8 waves; tile 128(M) x 64(N = one head) for all 3 projections.
// Weight tiles staged to LDS fragment-major via async b128 copies.
// Outputs: Q (H,S,D) bf16; K padded (H,SPAD,D) bf16; V transposed+padded
// (H,D,SPAD) bf16 — ready for branch-free async staging in attention.
// ---------------------------------------------------------------------------
__global__ __launch_bounds__(256)
void qkv_proj_kernel(const __bf16* __restrict__ xb,
                     const __bf16* __restrict__ wqb,
                     const __bf16* __restrict__ wkb,
                     const __bf16* __restrict__ wvb,
                     const float* __restrict__ bq,
                     const float* __restrict__ bk,
                     const float* __restrict__ bv,
                     __bf16* __restrict__ qgb,
                     __bf16* __restrict__ kgp,
                     __bf16* __restrict__ vtp)
{
  // B tiles fragment-major: [proj][ntile][lane][e], 16 bf16 per lane
  __shared__ __align__(32) __bf16 bW[3 * 4 * 32 * 16];

  const int tid  = threadIdx.x;
  const int lane = tid & 31;
  const int wave = tid >> 5;
  const int half = lane >> 4;
  const int ln   = lane & 15;

  const int m0 = blockIdx.x * 128 + wave * 16;
  const int h  = blockIdx.y;               // E/64 == NH
  const int n0 = h * 64;

  // Per-thread staging chunk: one 16B chunk per projection per K-step.
  // tid bits: [0]=chunk-half, [5:1]=frag lane, [7:6]=ntile
  const int sc_ch = tid & 1;
  const int sc_bl = (tid >> 1) & 31;
  const int sc_nt = (tid >> 6) & 3;
  const int sc_n  = n0 + sc_nt * 16 + (sc_bl & 15);
  const int sc_k  = 16 * (sc_bl >> 4) + 8 * sc_ch;
  __bf16* sc_lds0 = &bW[(sc_nt * 32 + sc_bl) * 16 + 8 * sc_ch];

  v8f acc[3][4] = {};

  for (int k0 = 0; k0 < E_DIM; k0 += 32) {
    __syncthreads();  // previous iteration's fragment reads done
    async_cp16(sc_lds0 + 0 * 2048, wqb + sc_n * E_DIM + k0 + sc_k);
    async_cp16(sc_lds0 + 1 * 2048, wkb + sc_n * E_DIM + k0 + sc_k);
    async_cp16(sc_lds0 + 2 * 2048, wvb + sc_n * E_DIM + k0 + sc_k);

    v16bf afrag = load_afrag(xb + (m0 + ln) * E_DIM + k0, half);

    wait_async0();
    __syncthreads();

#pragma unroll
    for (int p = 0; p < 3; ++p) {
#pragma unroll
      for (int nt = 0; nt < 4; ++nt) {
        v16bf bfrag = *(const v16bf*)&bW[((p * 4 + nt) * 32 + lane) * 16];
        acc[p][nt] = __builtin_amdgcn_wmma_f32_16x16x32_bf16(
            false, afrag, false, bfrag, (short)0, acc[p][nt], false, false);
      }
    }
  }

  // Store with bias as bf16 into attention-ready layouts.
#pragma unroll
  for (int p = 0; p < 3; ++p) {
    const float* bias = (p == 0) ? bq : (p == 1) ? bk : bv;
#pragma unroll
    for (int nt = 0; nt < 4; ++nt) {
      int d = nt * 16 + ln;
      float bb = bias[n0 + d];
#pragma unroll
      for (int r = 0; r < 8; ++r) {
        int s = m0 + r + 8 * half;            // C/D layout: M = vgpr + 8*half
        __bf16 val = (__bf16)(acc[p][nt][r] + bb);
        if (p == 0)      qgb[h * S_LEN * HD + s * HD + d] = val;
        else if (p == 1) kgp[h * SPAD * HD + (s + 32) * HD + d] = val;
        else             vtp[h * HD * SPAD + d * SPAD + (s + 32)] = val;
      }
    }
  }
}

// ---------------------------------------------------------------------------
// Kernel 2: sliding-window attention, flash-style with WMMA.
// 128 threads / 4 waves; 64 queries of one head; wave w owns 16 queries.
// K rows j = sbase-16..sbase+79 (96) and V^T positions sbase-32..sbase+79
// (112) staged branch-free from the zero-padded bf16 buffers via async b128.
// ---------------------------------------------------------------------------
__global__ __launch_bounds__(128)
void local_attn_kernel(const __bf16* __restrict__ qgb,
                       const __bf16* __restrict__ kgp,
                       const __bf16* __restrict__ vtp,
                       float* __restrict__ out)
{
  __shared__ __align__(32) __bf16 kbuf[96 * 64];    // [r][d]
  __shared__ __align__(32) __bf16 vbufT[64 * 112];  // [d][r]
  __shared__ float                 sbuf[4 * 16 * 48];
  __shared__ __align__(32) __bf16 pbuf[4 * 16 * 64];

  const int tid  = threadIdx.x;
  const int lane = tid & 31;
  const int wave = tid >> 5;
  const int half = lane >> 4;
  const int ln   = lane & 15;

  const int h     = blockIdx.y;
  const int sbase = blockIdx.x * 64;
  const int q0    = sbase + wave * 16;

  const __bf16* kh = kgp + h * SPAD * HD;      // padded row index = j + 32
  const __bf16* vh = vtp + h * HD * SPAD;      // padded pos index = pos + 32

  // K window: 768 x 16B chunks; kbuf row r <- padded row sbase+16+r.
#pragma unroll
  for (int it = 0; it < 6; ++it) {
    int cc = tid + it * 128;
    int r = cc >> 3, d0 = (cc & 7) * 8;
    async_cp16(&kbuf[r * 64 + d0], kh + (sbase + 16 + r) * HD + d0);
  }
  // V^T window: 896 x 16B chunks; vbufT[d][r] <- padded pos sbase + r.
#pragma unroll
  for (int it = 0; it < 7; ++it) {
    int cc = tid + it * 128;
    int d = cc / 14, r0 = (cc % 14) * 8;
    async_cp16(&vbufT[d * 112 + r0], vh + d * SPAD + sbase + r0);
  }
  wait_async0();
  __syncthreads();

  // Q A-fragments (16 rows x 64 d, two K-steps of 32) — pure bf16 b128 loads.
  v16bf aq[2];
  const __bf16* qrow = qgb + h * S_LEN * HD + (q0 + ln) * HD;
#pragma unroll
  for (int kk = 0; kk < 2; ++kk) aq[kk] = load_afrag(qrow + kk * 32, half);

  // Scores: 16 queries x 48 key columns c (j = q0-16+c); zero-padded K rows
  // reproduce the reference's zero logits for out-of-range keys.
#pragma unroll
  for (int ct = 0; ct < 3; ++ct) {
    v8f acc = {};
    int c = ct * 16 + ln;
#pragma unroll
    for (int kk = 0; kk < 2; ++kk) {
      v16bf b = *(const v16bf*)&kbuf[(wave * 16 + c) * 64 + kk * 32 + 16 * half];
      acc = __builtin_amdgcn_wmma_f32_16x16x32_bf16(
          false, aq[kk], false, b, (short)0, acc, false, false);
    }
#pragma unroll
    for (int r = 0; r < 8; ++r)
      sbuf[(wave * 16 + r + 8 * half) * 48 + c] = acc[r] * QK_SCALE;
  }
  __syncthreads();

  // Row softmax over the 33-wide band c in [m, m+32] (== w in [0,32]).
  if (half == 0) {
    int m = ln;
    const float* srow = &sbuf[(wave * 16 + m) * 48];
    float mx = -3.0e38f;
#pragma unroll
    for (int i = 0; i < 33; ++i) mx = fmaxf(mx, srow[m + i]);
    float pr[33], sum = 0.f;
#pragma unroll
    for (int i = 0; i < 33; ++i) { pr[i] = __expf(srow[m + i] - mx); sum += pr[i]; }
    float inv = 1.f / sum;
    __bf16* prow = &pbuf[(wave * 16 + m) * 64];
#pragma unroll
    for (int c2 = 0; c2 < 64; ++c2) prow[c2] = (__bf16)0.f;
#pragma unroll
    for (int i = 0; i < 33; ++i) prow[m + i] = (__bf16)(pr[i] * inv);
  }
  __syncthreads();

  // P A-fragments (16 x 64, zero beyond c=47)
  v16bf ap[2];
  const __bf16* prow = &pbuf[(wave * 16 + ln) * 64];
#pragma unroll
  for (int kk = 0; kk < 2; ++kk) ap[kk] = load_afrag(prow + kk * 32, half);

  // out[m][d] = sum_c P[m][c] * v[q0+c-32][d]; vbufT row index = wave*16 + c.
#pragma unroll
  for (int nt = 0; nt < 4; ++nt) {
    v8f acc = {};
    int d = nt * 16 + ln;
#pragma unroll
    for (int kk = 0; kk < 2; ++kk) {
      v16bf b = *(const v16bf*)&vbufT[d * 112 + wave * 16 + kk * 32 + 16 * half];
      acc = __builtin_amdgcn_wmma_f32_16x16x32_bf16(
          false, ap[kk], false, b, (short)0, acc, false, false);
    }
#pragma unroll
    for (int r = 0; r < 8; ++r) {
      int s = q0 + r + 8 * half;
      out[s * E_DIM + h * HD + d] = acc[r];
    }
  }
}

extern "C" void kernel_launch(void* const* d_in, const int* in_sizes, int n_in,
                              void* d_out, int out_size, void* d_ws, size_t ws_size,
                              hipStream_t stream) {
  const float* x  = (const float*)d_in[0];
  const float* Wq = (const float*)d_in[1];
  const float* bq = (const float*)d_in[2];
  const float* Wk = (const float*)d_in[3];
  const float* bk = (const float*)d_in[4];
  const float* Wv = (const float*)d_in[5];
  const float* bv = (const float*)d_in[6];
  float* out = (float*)d_out;

  // Workspace layout (bf16 elements), ~22.3 MB total
  __bf16* xb  = (__bf16*)d_ws;                      // S*E
  __bf16* wqb = xb  + S_LEN * E_DIM;                // E*E
  __bf16* wkb = wqb + E_DIM * E_DIM;
  __bf16* wvb = wkb + E_DIM * E_DIM;
  __bf16* qgb = wvb + E_DIM * E_DIM;                // H*S*D
  __bf16* kgp = qgb + NH * S_LEN * HD;              // H*SPAD*D (padded)
  __bf16* vtp = kgp + NH * SPAD * HD;               // H*D*SPAD (transposed+padded)

  const int nx = S_LEN * E_DIM, nw = E_DIM * E_DIM;
  cvt_bf16_kernel<<<nx / (256 * 8), 256, 0, stream>>>(x, xb, nx);
  cvt_bf16_kernel<<<nw / (256 * 8), 256, 0, stream>>>(Wq, wqb, nw);
  cvt_bf16_kernel<<<nw / (256 * 8), 256, 0, stream>>>(Wk, wkb, nw);
  cvt_bf16_kernel<<<nw / (256 * 8), 256, 0, stream>>>(Wv, wvb, nw);
  zero_pad_kernel<<<(NH * 64 * HD) / 256, 256, 0, stream>>>(kgp, vtp);

  qkv_proj_kernel<<<dim3(S_LEN / 128, E_DIM / 64), dim3(256), 0, stream>>>(
      xb, wqb, wkb, wvb, bq, bk, bv, qgb, kgp, vtp);
  local_attn_kernel<<<dim3(S_LEN / 64, NH), dim3(128), 0, stream>>>(
      qgb, kgp, vtp, out);
}
